// EfficientCrossScanSS2D_18519898981119
// MI455X (gfx1250) — matrix operand order repaired and checked by reading, not compiled, not run on hardware
//
#include <hip/hip_runtime.h>
#include <hip/hip_bf16.h>

typedef __attribute__((ext_vector_type(16))) _Float16 v16h;
typedef __attribute__((ext_vector_type(8)))  float    v8f;

#define M_TOK   32768      // B_SZ * H * W = 8*64*64
#define LSEQ    4096
#define DIN     80
#define DST     24
#define NCHUNK  64
#define CHL     64

// ---------------- WMMA fragment helpers (16x16x32 f16, wave32) ----------------
// A-matrix 16x32: lane&15 = row M, lane>>4 selects K half (+8). Element j=2v+p:
//   K = (v&3)*2 + p + (v>=4?16:0) + hi*8  -> two contiguous 8-half runs.
__device__ __forceinline__ v16h load_fragA_row(const _Float16* row, int kbase, int hi) {
  v16h f;
  int o = kbase + (hi << 3);
#pragma unroll
  for (int v = 0; v < 8; ++v) {
    int k = o + ((v & 3) << 1) + ((v >= 4) ? 16 : 0);
    f[2 * v]     = row[k];
    f[2 * v + 1] = row[k + 1];
  }
  return f;
}

// B-matrix 32x16: lane&15 = column N, lane>>4 selects K half (+8);
// same K order as A: two contiguous 8-half runs per lane-half.
__device__ __forceinline__ v16h load_fragB_row(const _Float16* rowN, int kbase, int hi) {
  v16h f;
  int o = kbase + (hi << 3);
#pragma unroll
  for (int v = 0; v < 8; ++v) {
    int k = o + ((v & 3) << 1) + ((v >= 4) ? 16 : 0);
    f[2 * v]     = rowN[k];
    f[2 * v + 1] = rowN[k + 1];
  }
  return f;
}

__device__ __forceinline__ void store_tile(float* C, int ldc, int m0, int n0,
                                           const v8f& d, int lane) {
  int n = lane & 15, hi = lane >> 4;
#pragma unroll
  for (int r = 0; r < 8; ++r)
    C[(size_t)(m0 + r + hi * 8) * ldc + n0 + n] = d[r];
}

// ---------------- generic GEMM: C(MxN,f32) = A(MxK,f16) * W(NxK,f16)^T --------
__global__ void gemm_f16_wmma(const _Float16* __restrict__ A, int lda,
                              const _Float16* __restrict__ W, int ldw,
                              float* __restrict__ C, int ldc,
                              int Ntiles, int Ksteps) {
  int wid  = (blockIdx.x * blockDim.x + threadIdx.x) >> 5;
  int lane = threadIdx.x & 31;
  int mt = wid / Ntiles, nt = wid % Ntiles;
  int m0 = mt * 16, n0 = nt * 16;
  int hi = lane >> 4;
  const _Float16* Arow = A + (size_t)(m0 + (lane & 15)) * lda;
  const _Float16* Wrow = W + (size_t)(n0 + (lane & 15)) * ldw;
  v8f acc = {};
  for (int ks = 0; ks < Ksteps; ++ks) {
    v16h a = load_fragA_row(Arow, ks * 32, hi);
    v16h b = load_fragB_row(Wrow, ks * 32, hi);
    acc = __builtin_amdgcn_wmma_f32_16x16x32_f16(false, a, false, b,
                                                 (short)0, acc, false, false);
  }
  store_tile(C, ldc, m0, n0, acc, lane);
}

// ---------------- weight prep: f32 -> f16 (zero-padded K to 96) ---------------
__global__ void prep_weights(const float* __restrict__ ipw, const float* __restrict__ xpw,
                             const float* __restrict__ opw, const float* __restrict__ fuw,
                             _Float16* Wip, _Float16* Wxp, _Float16* Wop, _Float16* Wfu) {
  int idx = blockIdx.x * blockDim.x + threadIdx.x;
  if (idx < 160 * 64) Wip[idx] = (_Float16)ipw[idx];
  if (idx < 64 * 96) {
    int n = idx / 96, k = idx % 96;
    Wxp[idx] = (n < 52 && k < 80) ? (_Float16)xpw[n * 80 + k] : (_Float16)0.f;
    Wop[idx] = (k < 80) ? (_Float16)opw[n * 80 + k] : (_Float16)0.f;
  }
  if (idx < 64 * 64) Wfu[idx] = (_Float16)fuw[idx];
}

// ---------------- cross-scan gather + LayerNorm -> f16 seq --------------------
__global__ void cross_scan_ln(const float* __restrict__ x, const float* __restrict__ lnw,
                              const float* __restrict__ lnb, _Float16* __restrict__ seqh) {
  int t = blockIdx.x * blockDim.x + threadIdx.x;   // token = b*4096 + l
  int b = t >> 12, p = t & 4095;
  int pt = ((p & 63) << 6) + (p >> 6);             // transposed spatial index
  int q[4] = { p, 4095 - p, pt, 4095 - pt };
  const float* xb = x + ((size_t)b << 18);         // b * 64 * 4096
  float v[64], mu = 0.f;
#pragma unroll
  for (int c = 0; c < 64; ++c) {
    float val = xb[((size_t)c << 12) + q[c >> 4]];
    v[c] = val; mu += val;
  }
  mu *= (1.f / 64.f);
  float var = 0.f;
#pragma unroll
  for (int c = 0; c < 64; ++c) { float d = v[c] - mu; var += d * d; }
  float r = rsqrtf(var * (1.f / 64.f) + 1e-5f);
  _Float16* o = seqh + (size_t)t * 64;
#pragma unroll
  for (int c = 0; c < 64; ++c)
    o[c] = (_Float16)((v[c] - mu) * r * lnw[c] + lnb[c]);
}

// ---------------- depthwise causal conv(4) + SiLU -----------------------------
__global__ void conv_silu(const float* __restrict__ xz, const float* __restrict__ cw,
                          const float* __restrict__ cb,
                          float* __restrict__ xcf, _Float16* __restrict__ xch) {
  int idx = blockIdx.x * blockDim.x + threadIdx.x;   // M_TOK * 96
  int i = idx / 96, d = idx % 96;
  if (d >= DIN) { xch[(size_t)i * 96 + d] = (_Float16)0.f; return; }
  int l = i & 4095;
  const float* col = xz + (size_t)i * 160 + d;
  float acc = cb[d];
#pragma unroll
  for (int k = 0; k < 4; ++k) {
    int ll = l - 3 + k;
    if (ll >= 0) acc += col[(ll - l) * 160] * cw[d * 4 + k];
  }
  float s = acc / (1.f + __expf(-acc));              // SiLU
  xcf[(size_t)i * 80 + d] = s;
  xch[(size_t)i * 96 + d] = (_Float16)s;
}

// ---------------- dt = softplus(dt_in @ dt_proj_w^T + b) ----------------------
__global__ void dt_softplus(const float* __restrict__ xdbl, const float* __restrict__ dtw,
                            const float* __restrict__ dtb, float* __restrict__ dt) {
  int idx = blockIdx.x * blockDim.x + threadIdx.x;   // M_TOK * 80
  int i = idx / 80, d = idx % 80;
  const float* r = xdbl + (size_t)i * 64;
  float a = dtb[d];
#pragma unroll
  for (int rr = 0; rr < 4; ++rr) a += r[rr] * dtw[d * 4 + rr];
  dt[idx] = (a > 20.f) ? a : log1pf(__expf(a));
}

// ---------------- chunked selective scan --------------------------------------
// Phase 1: per (b,d,chunk) wave, lane = state s: local recurrence from h=0,
// emit per-chunk (prod dA, local h).
__global__ void scan_phase1(const float* __restrict__ dt, const float* __restrict__ xcf,
                            const float* __restrict__ xdbl, const float* __restrict__ A_log,
                            float* __restrict__ P, float* __restrict__ Hc) {
  int wid = (blockIdx.x * blockDim.x + threadIdx.x) >> 5;
  int s = threadIdx.x & 31;
  int ch = wid & 63, bd = wid >> 6;
  int d = bd % DIN, b = bd / DIN;
  float Ads = (s < DST) ? -__expf(A_log[d * DST + s]) : 0.f;
  size_t tok0 = ((size_t)b << 12) + (size_t)ch * CHL;
  float prodA = 1.f, h = 0.f;
  for (int l = 0; l < CHL; ++l) {
    size_t tk = tok0 + l;
    float dtv = dt[tk * 80 + d];
    float xv  = xcf[tk * 80 + d];
    float Bv  = (s < DST) ? xdbl[tk * 64 + 4 + s] : 0.f;
    float a = __expf(dtv * Ads);
    h = fmaf(a, h, dtv * Bv * xv);
    prodA *= a;
  }
  size_t o = ((size_t)wid << 5) + s;
  P[o] = prodA; Hc[o] = h;
}

// Phase 2: tiny sequential scan over 64 chunk summaries per (b,d,s).
__global__ void scan_phase2(const float* __restrict__ P, const float* __restrict__ Hc,
                            float* __restrict__ Hs) {
  int idx = blockIdx.x * blockDim.x + threadIdx.x;   // 640 * 32
  int s = idx & 31, bd = idx >> 5;
  float h = 0.f;
  for (int ch = 0; ch < NCHUNK; ++ch) {
    size_t o = (((size_t)bd * NCHUNK + ch) << 5) + s;
    Hs[o] = h;
    h = P[o] * h + Hc[o];
  }
}

// Phase 3: replay chunks with known start state, y_t = sum_s h*C via shfl-reduce.
__global__ void scan_phase3(const float* __restrict__ dt, const float* __restrict__ xcf,
                            const float* __restrict__ xdbl, const float* __restrict__ A_log,
                            const float* __restrict__ Hs, float* __restrict__ ys) {
  int wid = (blockIdx.x * blockDim.x + threadIdx.x) >> 5;
  int s = threadIdx.x & 31;
  int ch = wid & 63, bd = wid >> 6;
  int d = bd % DIN, b = bd / DIN;
  float Ads = (s < DST) ? -__expf(A_log[d * DST + s]) : 0.f;
  size_t tok0 = ((size_t)b << 12) + (size_t)ch * CHL;
  float h = Hs[((size_t)wid << 5) + s];
  for (int l = 0; l < CHL; ++l) {
    size_t tk = tok0 + l;
    float dtv = dt[tk * 80 + d];
    float xv  = xcf[tk * 80 + d];
    float Bv  = (s < DST) ? xdbl[tk * 64 + 4 + s] : 0.f;
    float Cv  = (s < DST) ? xdbl[tk * 64 + 28 + s] : 0.f;
    float a = __expf(dtv * Ads);
    h = fmaf(a, h, dtv * Bv * xv);
    float pr = h * Cv;
#pragma unroll
    for (int off = 16; off; off >>= 1) pr += __shfl_xor(pr, off, 32);
    if (s == 0) ys[tk * 80 + d] = pr;
  }
}

// ---------------- y = (ys + xc*D) * silu(z) -> f16 (padded) -------------------
__global__ void ydz(const float* __restrict__ ys, const float* __restrict__ xcf,
                    const float* __restrict__ xz, const float* __restrict__ Dp,
                    _Float16* __restrict__ yh) {
  int idx = blockIdx.x * blockDim.x + threadIdx.x;   // M_TOK * 96
  int i = idx / 96, d = idx % 96;
  if (d >= DIN) { yh[idx] = (_Float16)0.f; return; }
  float y = ys[(size_t)i * 80 + d] + xcf[(size_t)i * 80 + d] * Dp[d];
  float z = xz[(size_t)i * 160 + 80 + d];
  y *= z / (1.f + __expf(-z));
  yh[(size_t)i * 96 + d] = (_Float16)y;
}

// ---------------- inverse cross-scan gather + fusion GEMM + residual ----------
__global__ void fusion_wmma(const float* __restrict__ oseq, const _Float16* __restrict__ Wfu,
                            const float* __restrict__ x, const float* __restrict__ scale,
                            float* __restrict__ out) {
  int wid  = (blockIdx.x * blockDim.x + threadIdx.x) >> 5;
  int lane = threadIdx.x & 31;
  int mt = wid >> 2, nt = wid & 3;                   // Ntiles = 4
  int t0 = mt * 16, n0 = nt * 16;
  int m = lane & 15, hi = lane >> 4;
  int t = t0 + m;
  int b = t >> 12, p = t & 4095;
  int pt = ((p & 63) << 6) + (p >> 6);
  int lmap[4] = { p, 4095 - p, pt, 4095 - pt };
  const _Float16* Wrow = Wfu + (size_t)(n0 + (lane & 15)) * 64;
  v8f acc = {};
#pragma unroll
  for (int ks = 0; ks < 2; ++ks) {
    v16h a;
#pragma unroll
    for (int v = 0; v < 8; ++v) {
      int k = ks * 32 + (hi << 3) + ((v & 3) << 1) + ((v >= 4) ? 16 : 0);
      int dir = k >> 4;                              // k even -> same dir for k,k+1
      const float* src = oseq + (((size_t)(b << 12)) + lmap[dir]) * 64;
      a[2 * v]     = (_Float16)src[k];
      a[2 * v + 1] = (_Float16)src[k + 1];
    }
    v16h bfr = load_fragB_row(Wrow, ks * 32, hi);
    acc = __builtin_amdgcn_wmma_f32_16x16x32_f16(false, a, false, bfr,
                                                 (short)0, acc, false, false);
  }
  float sc = scale[0];
  int nn = n0 + (lane & 15);
#pragma unroll
  for (int r = 0; r < 8; ++r) {
    int tt = t0 + r + hi * 8;
    int bb = tt >> 12, pp = tt & 4095;
    size_t idx = (((size_t)(bb * 64 + nn)) << 12) + pp;
    out[idx] = x[idx] + sc * acc[r];
  }
}

// ---------------- host orchestration ------------------------------------------
extern "C" void kernel_launch(void* const* d_in, const int* in_sizes, int n_in,
                              void* d_out, int out_size, void* d_ws, size_t ws_size,
                              hipStream_t stream) {
  const float* x    = (const float*)d_in[0];
  const float* lnw  = (const float*)d_in[1];
  const float* lnb  = (const float*)d_in[2];
  const float* ipw  = (const float*)d_in[3];
  const float* cw   = (const float*)d_in[4];
  const float* cb   = (const float*)d_in[5];
  const float* xpw  = (const float*)d_in[6];
  const float* dtw  = (const float*)d_in[7];
  const float* dtbv = (const float*)d_in[8];
  const float* alog = (const float*)d_in[9];
  const float* Dp   = (const float*)d_in[10];
  const float* opw  = (const float*)d_in[11];
  const float* fuw  = (const float*)d_in[12];
  const float* scal = (const float*)d_in[13];
  float* out = (float*)d_out;

  char* w = (char*)d_ws;
  auto carve = [&](size_t bytes) -> void* {
    void* p = (void*)w; w += (bytes + 255) & ~(size_t)255; return p;
  };
  const size_t M = M_TOK;
  _Float16* Wip  = (_Float16*)carve(160 * 64 * 2);
  _Float16* Wxp  = (_Float16*)carve(64 * 96 * 2);
  _Float16* Wop  = (_Float16*)carve(64 * 96 * 2);
  _Float16* Wfu  = (_Float16*)carve(64 * 64 * 2);
  _Float16* seqh = (_Float16*)carve(M * 64 * 2);
  float*    xz   = (float*)   carve(M * 160 * 4);
  float*    xcf  = (float*)   carve(M * 80 * 4);
  _Float16* xch  = (_Float16*)carve(M * 96 * 2);
  float*    xdbl = (float*)   carve(M * 64 * 4);
  float*    dtb  = (float*)   carve(M * 80 * 4);
  float*    ysb  = (float*)   carve(M * 80 * 4);
  _Float16* yh   = (_Float16*)carve(M * 96 * 2);
  float*    oseq = (float*)   carve(M * 64 * 4);
  float*    P    = (float*)   carve(640 * 64 * 32 * 4);
  float*    Hc   = (float*)   carve(640 * 64 * 32 * 4);
  float*    Hs   = (float*)   carve(640 * 64 * 32 * 4);

  prep_weights<<<40, 256, 0, stream>>>(ipw, xpw, opw, fuw, Wip, Wxp, Wop, Wfu);
  cross_scan_ln<<<M / 256, 256, 0, stream>>>(x, lnw, lnb, seqh);
  // in_proj: M x 64 -> 160 : 2048*10 tiles / 8 waves per block
  gemm_f16_wmma<<<2560, 256, 0, stream>>>(seqh, 64, Wip, 64, xz, 160, 10, 2);
  conv_silu<<<(M * 96) / 256, 256, 0, stream>>>(xz, cw, cb, xcf, xch);
  // x_proj: M x 96(pad) -> 64(pad of 52)
  gemm_f16_wmma<<<1024, 256, 0, stream>>>(xch, 96, Wxp, 96, xdbl, 64, 4, 3);
  dt_softplus<<<(M * 80) / 256, 256, 0, stream>>>(xdbl, dtw, dtbv, dtb);
  scan_phase1<<<5120, 256, 0, stream>>>(dtb, xcf, xdbl, alog, P, Hc);
  scan_phase2<<<80, 256, 0, stream>>>(P, Hc, Hs);
  scan_phase3<<<5120, 256, 0, stream>>>(dtb, xcf, xdbl, alog, Hs, ysb);
  ydz<<<(M * 96) / 256, 256, 0, stream>>>(ysb, xcf, xz, Dp, yh);
  // out_proj: M x 96(pad) -> 64
  gemm_f16_wmma<<<1024, 256, 0, stream>>>(yh, 96, Wop, 96, oseq, 64, 4, 3);
  // inverse scan + fusion + residual
  fusion_wmma<<<1024, 256, 0, stream>>>(oseq, Wfu, x, scal, out);
}
